// EncoderBlock_30777735644082
// MI455X (gfx1250) — matrix operand order repaired and checked by reading, not compiled, not run on hardware
//
#include <hip/hip_runtime.h>
#include <hip/hip_bf16.h>

typedef __attribute__((ext_vector_type(16))) _Float16 v16h;
typedef __attribute__((ext_vector_type(8)))  float    v8f;

union Frag {
    v16h v;
    uint4 q[2];
};

#define BQ 2
#define SQ 2048
#define DQ 768
#define HQ 12
#define DKQ 64
#define DFFQ 3072
#define MS (BQ * SQ)   // 4096 token rows

// ---------------------------------------------------------------------------
// CDNA5 async global->LDS (ASYNCcnt-tracked DMA), with fallback
// ---------------------------------------------------------------------------
#if defined(__HIP_DEVICE_COMPILE__) && defined(__has_builtin)
# if __has_builtin(__builtin_amdgcn_global_load_async_to_lds_b128) && \
     __has_builtin(__builtin_amdgcn_s_wait_asynccnt)
#  define HAS_ASYNC 1
# endif
#endif
#ifndef HAS_ASYNC
# define HAS_ASYNC 0
#endif

// Exact parameter types per hipcc diagnostic: vector_size(16) int, AS1 / AS3
typedef int vint4 __attribute__((vector_size(4 * sizeof(int))));
typedef __attribute__((address_space(1))) vint4 g_vint4;
typedef __attribute__((address_space(3))) vint4 l_vint4;

__device__ __forceinline__ void async_cp16(void* l, const void* g) {
#if HAS_ASYNC
    // b128: each lane moves 16B global -> LDS via the async engine
    __builtin_amdgcn_global_load_async_to_lds_b128(
        (g_vint4*)(uintptr_t)g,
        (l_vint4*)(unsigned int)(uintptr_t)l,   // generic LDS addr low 32b = offset
        0, 0);
#else
    *(uint4*)l = *(const uint4*)g;
#endif
}

__device__ __forceinline__ void wait_async_3() {
#if HAS_ASYNC
    __builtin_amdgcn_s_wait_asynccnt(3);
#endif
}
__device__ __forceinline__ void wait_async_0() {
#if HAS_ASYNC
    __builtin_amdgcn_s_wait_asynccnt(0);
#endif
}

// ---------------------------------------------------------------------------
// Weight convert + transpose: w[K][N] fp32 -> wt[N][K] f16
// ---------------------------------------------------------------------------
__global__ __launch_bounds__(256) void conv_t_kernel(const float* __restrict__ w,
                                                     _Float16* __restrict__ wt,
                                                     int K, int N) {
    int idx = blockIdx.x * 256 + threadIdx.x;
    if (idx >= K * N) return;
    int k = idx / N, n = idx % N;
    wt[(size_t)n * K + k] = (_Float16)w[idx];
}

// ---------------------------------------------------------------------------
// V transpose: v[row][h*64+dk] f16 -> vt[(b*H+h)*64+dk][s] f16
// ---------------------------------------------------------------------------
__global__ __launch_bounds__(256) void vtrans_kernel(const _Float16* __restrict__ v,
                                                     _Float16* __restrict__ vt) {
    int idx = blockIdx.x * 256 + threadIdx.x;
    if (idx >= MS * DQ) return;
    int row = idx / DQ, col = idx % DQ;
    int b = row >> 11, s = row & 2047;
    int h = col >> 6, dk = col & 63;
    vt[((size_t)((b * HQ + h) * DKQ + dk)) * SQ + s] = v[idx];
}

// ---------------------------------------------------------------------------
// LayerNorm (unbiased std, eps added to std) -> f16 output
// ---------------------------------------------------------------------------
__global__ __launch_bounds__(256) void ln_kernel(const float* __restrict__ x,
                                                 const float* __restrict__ gamma,
                                                 const float* __restrict__ beta,
                                                 _Float16* __restrict__ out) {
    __shared__ float red0[256];
    __shared__ float red1[256];
    int row = blockIdx.x;
    const float* xr = x + (size_t)row * DQ;
    float v[3], s = 0.f, s2 = 0.f;
    for (int i = 0; i < 3; ++i) {
        v[i] = xr[threadIdx.x + i * 256];
        s += v[i];
        s2 += v[i] * v[i];
    }
    red0[threadIdx.x] = s;
    red1[threadIdx.x] = s2;
    __syncthreads();
    for (int off = 128; off > 0; off >>= 1) {
        if (threadIdx.x < off) {
            red0[threadIdx.x] += red0[threadIdx.x + off];
            red1[threadIdx.x] += red1[threadIdx.x + off];
        }
        __syncthreads();
    }
    float mean = red0[0] * (1.0f / DQ);
    float var = (red1[0] - (float)DQ * mean * mean) * (1.0f / (DQ - 1));
    float inv = 1.0f / (sqrtf(fmaxf(var, 0.f)) + 1e-6f);
    for (int i = 0; i < 3; ++i) {
        int c = threadIdx.x + i * 256;
        out[(size_t)row * DQ + c] = (_Float16)(gamma[c] * (v[i] - mean) * inv + beta[c]);
    }
}

// ---------------------------------------------------------------------------
// WMMA GEMM: out[M][N] = act(A[M][K] @ Wt[N][K]^T + bias) + resid
// 256 threads = 8 waves; block tile 128x64; K templated; K step 32.
// Double-buffered LDS, async global->LDS DMA overlapped with WMMA.
// ---------------------------------------------------------------------------
#define MT 128
#define NT 64
#define LDP 40  // padded LDS row stride in halfs (80B, 16B aligned)

template <int K>
__global__ __launch_bounds__(256) void gemm_wmma_kernel(
        const _Float16* __restrict__ A, const _Float16* __restrict__ Wt,
        const float* __restrict__ bias, const float* __restrict__ resid,
        float* __restrict__ outF, _Float16* __restrict__ outH,
        int N, int relu) {
    __shared__ __align__(16) _Float16 la[2][MT][LDP];
    __shared__ __align__(16) _Float16 lw[2][NT][LDP];
    constexpr int NK = K / 32;

    int tid = threadIdx.x;
    int lane = tid & 31, wave = tid >> 5;
    int ln = lane & 15, hi = lane >> 4;
    int m0 = blockIdx.y * MT, n0 = blockIdx.x * NT;

    // per-thread staging chunk coordinates (8 halfs = 16B per chunk)
    int ar = tid >> 2;           // 0..63  (also handles row ar+64)
    int ac = (tid & 3) * 8;      // 0,8,16,24
    const _Float16* gA = A + (size_t)(m0 + ar) * K + ac;
    const _Float16* gW = Wt + (size_t)(n0 + ar) * K + ac;

    auto issue = [&](int s, int buf) {
        const _Float16* ga = gA + s * 32;
        async_cp16(&la[buf][ar][ac], ga);
        async_cp16(&la[buf][ar + 64][ac], ga + (size_t)64 * K);
        async_cp16(&lw[buf][ar][ac], gW + s * 32);
    };

    v8f acc[4] = {};
    issue(0, 0);
    for (int s = 0; s < NK; ++s) {
        int buf = s & 1;
        if (s + 1 < NK) {
            issue(s + 1, buf ^ 1);
            wait_async_3();      // stage s complete; stage s+1 in flight
        } else {
            wait_async_0();
        }
        __syncthreads();
        // A fragment: m = ln, k = (i&7) + (i>=8?16:0) + hi*8
        Frag fa;
        int off8 = hi * 8;
        fa.q[0] = *(const uint4*)&la[buf][wave * 16 + ln][off8];
        fa.q[1] = *(const uint4*)&la[buf][wave * 16 + ln][16 + off8];
#pragma unroll
        for (int nt = 0; nt < 4; ++nt) {
            // B fragment: n = ln, k = (i&7) + (i>=8?16:0) + hi*8
            Frag fb;
            int kb = hi * 8;
            fb.q[0] = *(const uint4*)&lw[buf][nt * 16 + ln][kb];
            fb.q[1] = *(const uint4*)&lw[buf][nt * 16 + ln][kb + 16];
            acc[nt] = __builtin_amdgcn_wmma_f32_16x16x32_f16(
                false, fa.v, false, fb.v, (short)0, acc[nt], false, false);
        }
        __syncthreads();   // all waves done reading buf before stage s+2 overwrites it
    }
    // Epilogue: C layout m = j + 8*hi, n = ln
#pragma unroll
    for (int nt = 0; nt < 4; ++nt) {
        int n = n0 + nt * 16 + ln;
        float bv = bias ? bias[n] : 0.f;
#pragma unroll
        for (int j = 0; j < 8; ++j) {
            int m = m0 + wave * 16 + j + hi * 8;
            float val = acc[nt][j] + bv;
            if (relu) val = val > 0.f ? val : 0.f;
            if (resid) val += resid[(size_t)m * N + n];
            if (outF) outF[(size_t)m * N + n] = val;
            if (outH) outH[(size_t)m * N + n] = (_Float16)val;
        }
    }
}

// ---------------------------------------------------------------------------
// Flash attention: one wave per 16-query tile. 4 waves/block.
// Q,K: f16 [MS][768] (col h*64+dk). Vt: f16 [(b*H+h)*64+dk][S].
// ctx out: f16 [MS][768].
// ---------------------------------------------------------------------------
__global__ __launch_bounds__(128) void attn_kernel(
        const _Float16* __restrict__ Q, const _Float16* __restrict__ Kx,
        const _Float16* __restrict__ Vt, const int* __restrict__ mask,
        _Float16* __restrict__ ctx) {
    __shared__ __align__(16) _Float16 lp[4][16][32];
    int tid = threadIdx.x, lane = tid & 31, wv = tid >> 5;
    int w = blockIdx.x * 4 + wv;            // 0..3071
    int qt = w & 127;                        // S/16 = 128 query tiles
    int h = (w >> 7) % HQ;
    int b = (w >> 7) / HQ;
    int ln = lane & 15, hi = lane >> 4;
    int off8 = hi * 8;

    // Q fragments (2 x 16x32 over DK=64), resident in registers
    Frag fq[2];
    {
        const _Float16* qp = Q + (size_t)(b * SQ + qt * 16 + ln) * DQ + h * DKQ;
        fq[0].q[0] = *(const uint4*)(qp + off8);
        fq[0].q[1] = *(const uint4*)(qp + 16 + off8);
        fq[1].q[0] = *(const uint4*)(qp + 32 + off8);
        fq[1].q[1] = *(const uint4*)(qp + 48 + off8);
    }

    float mrun[8], lrun[8];
    v8f acc[4] = {};
    for (int j = 0; j < 8; ++j) { mrun[j] = -1e30f; lrun[j] = 0.f; }

    const _Float16* kbase = Kx + (size_t)(b * SQ) * DQ + h * DKQ;
    const _Float16* vtb = Vt + (size_t)((b * HQ + h) * DKQ) * SQ;
    const int* mk = mask + b * SQ;

    for (int kt = 0; kt < SQ; kt += 32) {
        // --- scores for 32 keys: two 16x16 tiles, each = 2 WMMAs over DK ---
        v8f s0 = {}, s1 = {};
        {
            const _Float16* kp = kbase + (size_t)(kt + ln) * DQ + hi * 8;
            Frag fb;
            fb.q[0] = *(const uint4*)(kp);
            fb.q[1] = *(const uint4*)(kp + 16);
            s0 = __builtin_amdgcn_wmma_f32_16x16x32_f16(false, fq[0].v, false, fb.v, (short)0, s0, false, false);
            fb.q[0] = *(const uint4*)(kp + 32);
            fb.q[1] = *(const uint4*)(kp + 48);
            s0 = __builtin_amdgcn_wmma_f32_16x16x32_f16(false, fq[1].v, false, fb.v, (short)0, s0, false, false);
        }
        {
            const _Float16* kp = kbase + (size_t)(kt + 16 + ln) * DQ + hi * 8;
            Frag fb;
            fb.q[0] = *(const uint4*)(kp);
            fb.q[1] = *(const uint4*)(kp + 16);
            s1 = __builtin_amdgcn_wmma_f32_16x16x32_f16(false, fq[0].v, false, fb.v, (short)0, s1, false, false);
            fb.q[0] = *(const uint4*)(kp + 32);
            fb.q[1] = *(const uint4*)(kp + 48);
            s1 = __builtin_amdgcn_wmma_f32_16x16x32_f16(false, fq[1].v, false, fb.v, (short)0, s1, false, false);
        }
        // --- mask, online softmax update ---
        int z0 = mk[kt + ln];
        int z1 = mk[kt + 16 + ln];
        float p0[8], p1[8];
#pragma unroll
        for (int j = 0; j < 8; ++j) {
            float a0 = z0 ? s0[j] * 0.125f : -1e9f;   // 1/sqrt(64)
            float a1 = z1 ? s1[j] * 0.125f : -1e9f;
            float t = fmaxf(a0, a1);
            for (int off = 1; off < 16; off <<= 1) t = fmaxf(t, __shfl_xor(t, off, 16));
            float mnew = fmaxf(mrun[j], t);
            float corr = __expf(mrun[j] - mnew);
            float e0 = __expf(a0 - mnew);
            float e1 = __expf(a1 - mnew);
            float rs = e0 + e1;
            for (int off = 1; off < 16; off <<= 1) rs += __shfl_xor(rs, off, 16);
            lrun[j] = lrun[j] * corr + rs;
            mrun[j] = mnew;
            p0[j] = e0; p1[j] = e1;
            for (int nt = 0; nt < 4; ++nt) acc[nt][j] *= corr;
        }
        // --- stage P (16x32 f16) through LDS to build the A fragment ---
#pragma unroll
        for (int j = 0; j < 8; ++j) {
            lp[wv][j + hi * 8][ln] = (_Float16)p0[j];
            lp[wv][j + hi * 8][16 + ln] = (_Float16)p1[j];
        }
        __syncthreads();
        Frag fp;
        fp.q[0] = *(const uint4*)&lp[wv][ln][off8];
        fp.q[1] = *(const uint4*)&lp[wv][ln][16 + off8];
        __syncthreads();
        // --- ctx += P @ V (4 WMMAs over DK=64) ---
#pragma unroll
        for (int nt = 0; nt < 4; ++nt) {
            const _Float16* vp = vtb + (size_t)(nt * 16 + ln) * SQ + kt + hi * 8;
            Frag fv;
            fv.q[0] = *(const uint4*)(vp);
            fv.q[1] = *(const uint4*)(vp + 16);
            acc[nt] = __builtin_amdgcn_wmma_f32_16x16x32_f16(
                false, fp.v, false, fv.v, (short)0, acc[nt], false, false);
        }
    }
    // --- normalize & write ctx (f16, layout [MS][768]) ---
#pragma unroll
    for (int nt = 0; nt < 4; ++nt)
#pragma unroll
        for (int j = 0; j < 8; ++j) {
            int m = b * SQ + qt * 16 + j + hi * 8;
            ctx[(size_t)m * DQ + h * DKQ + nt * 16 + ln] =
                (_Float16)(acc[nt][j] / lrun[j]);
        }
}

// ---------------------------------------------------------------------------
// Host-side orchestration
// ---------------------------------------------------------------------------
extern "C" void kernel_launch(void* const* d_in, const int* in_sizes, int n_in,
                              void* d_out, int out_size, void* d_ws, size_t ws_size,
                              hipStream_t stream) {
    const float* src    = (const float*)d_in[0];
    const int*   smask  = (const int*)d_in[1];
    const float* w_q    = (const float*)d_in[2];
    const float* w_k    = (const float*)d_in[3];
    const float* w_v    = (const float*)d_in[4];
    const float* w_o    = (const float*)d_in[5];
    const float* w_ff1  = (const float*)d_in[6];
    const float* b_ff1  = (const float*)d_in[7];
    const float* w_ff2  = (const float*)d_in[8];
    const float* b_ff2  = (const float*)d_in[9];
    const float* gamma1 = (const float*)d_in[10];
    const float* beta1  = (const float*)d_in[11];
    const float* gamma2 = (const float*)d_in[12];
    const float* beta2  = (const float*)d_in[13];
    float* out = (float*)d_out;

    char* p = (char*)d_ws;
    auto carve = [&](size_t bytes) -> char* {
        char* r = p;
        p += (bytes + 255) & ~(size_t)255;
        return r;
    };
    _Float16* x16  = (_Float16*)carve((size_t)MS * DQ * 2);
    _Float16* q16  = (_Float16*)carve((size_t)MS * DQ * 2);
    _Float16* k16  = (_Float16*)carve((size_t)MS * DQ * 2);
    _Float16* v16  = (_Float16*)carve((size_t)MS * DQ * 2);
    _Float16* vt16 = (_Float16*)carve((size_t)MS * DQ * 2);
    _Float16* c16  = (_Float16*)carve((size_t)MS * DQ * 2);
    _Float16* y16  = (_Float16*)carve((size_t)MS * DQ * 2);
    _Float16* h16  = (_Float16*)carve((size_t)MS * DFFQ * 2);
    float*    src2 = (float*)carve((size_t)MS * DQ * 4);
    _Float16* wqT  = (_Float16*)carve((size_t)DQ * DQ * 2);
    _Float16* wkT  = (_Float16*)carve((size_t)DQ * DQ * 2);
    _Float16* wvT  = (_Float16*)carve((size_t)DQ * DQ * 2);
    _Float16* woT  = (_Float16*)carve((size_t)DQ * DQ * 2);
    _Float16* wf1T = (_Float16*)carve((size_t)DQ * DFFQ * 2);
    _Float16* wf2T = (_Float16*)carve((size_t)DFFQ * DQ * 2);

    // 1. Convert + transpose weights to f16 [N][K]
    int nDD = DQ * DQ, nDF = DQ * DFFQ;
    conv_t_kernel<<<(nDD + 255) / 256, 256, 0, stream>>>(w_q, wqT, DQ, DQ);
    conv_t_kernel<<<(nDD + 255) / 256, 256, 0, stream>>>(w_k, wkT, DQ, DQ);
    conv_t_kernel<<<(nDD + 255) / 256, 256, 0, stream>>>(w_v, wvT, DQ, DQ);
    conv_t_kernel<<<(nDD + 255) / 256, 256, 0, stream>>>(w_o, woT, DQ, DQ);
    conv_t_kernel<<<(nDF + 255) / 256, 256, 0, stream>>>(w_ff1, wf1T, DQ, DFFQ);
    conv_t_kernel<<<(nDF + 255) / 256, 256, 0, stream>>>(w_ff2, wf2T, DFFQ, DQ);

    // 2. LN1 -> x16
    ln_kernel<<<MS, 256, 0, stream>>>(src, gamma1, beta1, x16);

    // 3. Q, K, V projections (f16 out)
    dim3 g768(DQ / NT, MS / MT);     // (12, 32)
    gemm_wmma_kernel<DQ><<<g768, 256, 0, stream>>>(x16, wqT, nullptr, nullptr,
                                                   nullptr, q16, DQ, 0);
    gemm_wmma_kernel<DQ><<<g768, 256, 0, stream>>>(x16, wkT, nullptr, nullptr,
                                                   nullptr, k16, DQ, 0);
    gemm_wmma_kernel<DQ><<<g768, 256, 0, stream>>>(x16, wvT, nullptr, nullptr,
                                                   nullptr, v16, DQ, 0);

    // 4. Transpose V for ctx-WMMA B operand
    vtrans_kernel<<<(MS * DQ + 255) / 256, 256, 0, stream>>>(v16, vt16);

    // 5. Flash attention -> c16
    attn_kernel<<<(BQ * HQ * (SQ / 16)) / 4, 128, 0, stream>>>(q16, k16, vt16,
                                                               smask, c16);

    // 6. Output projection + residual (f32) -> src2
    gemm_wmma_kernel<DQ><<<g768, 256, 0, stream>>>(c16, woT, nullptr, src,
                                                   src2, nullptr, DQ, 0);

    // 7. LN2 -> y16
    ln_kernel<<<MS, 256, 0, stream>>>(src2, gamma2, beta2, y16);

    // 8. FFN1: relu(y @ w_ff1 + b1) -> h16
    dim3 g3072(DFFQ / NT, MS / MT);  // (48, 32)
    gemm_wmma_kernel<DQ><<<g3072, 256, 0, stream>>>(y16, wf1T, b_ff1, nullptr,
                                                    nullptr, h16, DFFQ, 1);

    // 9. FFN2: h @ w_ff2 + b2 + src2 -> out (fp32)
    gemm_wmma_kernel<DFFQ><<<g768, 256, 0, stream>>>(h16, wf2T, b_ff2, src2,
                                                     out, nullptr, DQ, 0);
}
